// EngramV2_29703993819795
// MI455X (gfx1250) — matrix operand order repaired and checked by reading, not compiled, not run on hardware
//
#include <hip/hip_runtime.h>
#include <math.h>

#define NT      8192      // B*T
#define DD      1024      // D
#define TT      2048      // T
#define NGRAM_  3
#define BRANCH_ 4
#define BUCKET_ 100000

typedef __attribute__((ext_vector_type(16))) __bf16       v16bf;
typedef __attribute__((ext_vector_type(8)))  float        v8f;
typedef __attribute__((ext_vector_type(4)))  unsigned int v4u;
typedef __attribute__((ext_vector_type(8)))  int          v8i;
typedef __attribute__((ext_vector_type(4)))  int          v4i;

#if defined(__has_builtin)
#if __has_builtin(__builtin_amdgcn_tensor_load_to_lds)
#define USE_TDM 1
#endif
#endif
#ifndef USE_TDM
#define USE_TDM 0
#endif

__device__ __forceinline__ float warp_sum(float v) {
#pragma unroll
    for (int off = 16; off > 0; off >>= 1) v += __shfl_xor(v, off, 32);
    return v;
}

// ---------------------------------------------------------------------------
// f32 -> bf16 bulk convert (4 elems / thread)
// ---------------------------------------------------------------------------
__global__ __launch_bounds__(256)
void cvt_bf16_kernel(const float* __restrict__ in, __bf16* __restrict__ out) {
    const size_t i = ((size_t)blockIdx.x * 256 + threadIdx.x) * 4;
    float4 v = *(const float4*)(const void*)(in + i);
    out[i + 0] = (__bf16)v.x;
    out[i + 1] = (__bf16)v.y;
    out[i + 2] = (__bf16)v.z;
    out[i + 3] = (__bf16)v.w;
}

// ---------------------------------------------------------------------------
// LayerNorm of clipped hidden -> bf16   (one block per token, 256 thr)
// ---------------------------------------------------------------------------
__global__ __launch_bounds__(256)
void ln_clip_kernel(const float* __restrict__ x, __bf16* __restrict__ out) {
    const int tok = blockIdx.x;
    const float* row = x + (size_t)tok * DD;
    float vals[4], s = 0.f, s2 = 0.f;
#pragma unroll
    for (int i = 0; i < 4; ++i) {
        float v = row[threadIdx.x + i * 256];
        v = fminf(10.f, fmaxf(-10.f, v));
        vals[i] = v; s += v; s2 += v * v;
    }
    __shared__ float rs[8], rq[8];
    s = warp_sum(s); s2 = warp_sum(s2);
    const int lane = threadIdx.x & 31, w = threadIdx.x >> 5;
    if (lane == 0) { rs[w] = s; rq[w] = s2; }
    __syncthreads();
    if (w == 0) {
        float a = (lane < 8) ? rs[lane] : 0.f;
        float b = (lane < 8) ? rq[lane] : 0.f;
        a = warp_sum(a); b = warp_sum(b);
        if (lane == 0) { rs[0] = a; rq[0] = b; }
    }
    __syncthreads();
    const float mu = rs[0] * (1.f / DD);
    const float var = rq[0] * (1.f / DD) - mu * mu;
    const float r = rsqrtf(var + 1e-5f);
#pragma unroll
    for (int i = 0; i < 4; ++i)
        out[(size_t)tok * DD + threadIdx.x + i * 256] = (__bf16)((vals[i] - mu) * r);
}

// ---------------------------------------------------------------------------
// Gather table row by bucket, LayerNorm -> bf16
// ---------------------------------------------------------------------------
__global__ __launch_bounds__(256)
void gather_ln_kernel(const float* __restrict__ table_n, const int* __restrict__ bkt,
                      __bf16* __restrict__ out) {
    const int tok = blockIdx.x;
    const float* row = table_n + (size_t)bkt[tok] * DD;
    float vals[4], s = 0.f, s2 = 0.f;
#pragma unroll
    for (int i = 0; i < 4; ++i) {
        float v = row[threadIdx.x + i * 256];
        vals[i] = v; s += v; s2 += v * v;
    }
    __shared__ float rs[8], rq[8];
    s = warp_sum(s); s2 = warp_sum(s2);
    const int lane = threadIdx.x & 31, w = threadIdx.x >> 5;
    if (lane == 0) { rs[w] = s; rq[w] = s2; }
    __syncthreads();
    if (w == 0) {
        float a = (lane < 8) ? rs[lane] : 0.f;
        float b = (lane < 8) ? rq[lane] : 0.f;
        a = warp_sum(a); b = warp_sum(b);
        if (lane == 0) { rs[0] = a; rq[0] = b; }
    }
    __syncthreads();
    const float mu = rs[0] * (1.f / DD);
    const float var = rq[0] * (1.f / DD) - mu * mu;
    const float r = rsqrtf(var + 1e-5f);
#pragma unroll
    for (int i = 0; i < 4; ++i)
        out[(size_t)tok * DD + threadIdx.x + i * 256] = (__bf16)((vals[i] - mu) * r);
}

// ---------------------------------------------------------------------------
// TDM: load a 128x32 bf16 tile (row stride DD elems) into LDS, padded rows
// (16 DWORDs data + 4 DWORDs pad -> 40-half LDS pitch). ISA 08 §8 descriptor.
// ---------------------------------------------------------------------------
#if USE_TDM
__device__ __forceinline__ void tdm_load_tile(const void* gtile, unsigned lds_addr) {
    const unsigned long long ga = (unsigned long long)(uintptr_t)gtile;
    v4u g0 = { 1u,                                   // count=1, user descriptor
               lds_addr,                             // LDS byte address
               (unsigned)ga,                         // global_addr[31:0]
               (unsigned)(((ga >> 32) & 0x1FFFFFFull) | (2u << 30)) }; // [56:32]|type=2
    v8i g1;
    g1[0] = (1 << 16)     // data_size = 2 bytes
          | (1 << 20)     // pad_enable
          | (3 << 22)     // pad_interval = 16 DWORDs (64B tile row)
          | (3 << 25);    // pad_amount   = 4 DWORDs (16B pad -> 80B pitch)
    g1[1] = (int)((DD & 0xFFFFu) << 16);                      // tensor_dim0 lo16
    g1[2] = (int)((DD >> 16) | ((1048576u & 0xFFFFu) << 16)); // dim0 hi | dim1 lo
    g1[3] = (int)((1048576u >> 16) | (32u << 16));            // dim1 hi | tile_dim0=32
    g1[4] = 128;                                              // tile_dim1=128, tile_dim2=0
    g1[5] = DD;                                               // tensor_dim0_stride lo32
    g1[6] = 0;                                                // stride0 hi | stride1 lo
    g1[7] = 0;
    v4i z4 = {0, 0, 0, 0};
#if __clang_major__ >= 23
    v8i z8 = {0, 0, 0, 0, 0, 0, 0, 0};
    __builtin_amdgcn_tensor_load_to_lds(g0, g1, z4, z4, z8, 0);
#else
    __builtin_amdgcn_tensor_load_to_lds(g0, g1, z4, z4, 0);
#endif
}
#endif

#if !USE_TDM
__device__ __forceinline__ void stage_regs(const __bf16* __restrict__ A,
                                           const __bf16* __restrict__ Bp,
                                           size_t row0, size_t col0, int k0, int tid,
                                           uint4 aR[2], uint4 bR[2]) {
#pragma unroll
    for (int i = 0; i < 2; ++i) {
        int c = tid + i * 256;
        int r = c >> 2, kc = (c & 3) * 8;
        aR[i] = *(const uint4*)(const void*)(A + (row0 + r) * DD + k0 + kc);
        bR[i] = *(const uint4*)(const void*)(Bp + (col0 + r) * DD + k0 + kc);
    }
}
#endif

// ---------------------------------------------------------------------------
// C[z, m, o] = sum_d A[m, d] * B[z, o, d]   (A,B bf16 row-major, C f32)
// 128x128 tile / WG, 8 waves, wave tile 64x32 (8 wmma accums), 32 K-steps,
// double-buffered LDS fed by the Tensor Data Mover.
// ---------------------------------------------------------------------------
__global__ __launch_bounds__(256)
void gemm_bt_wmma(const __bf16* __restrict__ A, const __bf16* __restrict__ Bw,
                  float* __restrict__ C, size_t b_batch_stride, size_t c_batch_stride) {
    __shared__ __attribute__((aligned(16))) __bf16 sA[2][128][40];
    __shared__ __attribute__((aligned(16))) __bf16 sB[2][128][40];

    const int tid  = threadIdx.x;
    const int lane = tid & 31;
    const int wid  = tid >> 5;
    const int wm   = wid & 1;   // 2 waves along M (2*64 = 128)
    const int wn   = wid >> 1;  // 4 waves along N (4*32 = 128)

    const size_t row0 = (size_t)blockIdx.x * 128;
    const size_t col0 = (size_t)blockIdx.y * 128;
    const __bf16* Bp = Bw + (size_t)blockIdx.z * b_batch_stride;
    float*        Cp = C  + (size_t)blockIdx.z * c_batch_stride;

    v8f acc[4][2] = {};
    const int NK = DD / 32;   // 32 K-steps

#if USE_TDM
    const __bf16* Abase = A  + row0 * DD;
    const __bf16* Bbase = Bp + col0 * DD;
    if (tid == 0) {                       // exactly one descriptor pair per tile
        tdm_load_tile(Abase, (unsigned)(uintptr_t)&sA[0][0][0]);
        tdm_load_tile(Bbase, (unsigned)(uintptr_t)&sB[0][0][0]);
    }
    __builtin_amdgcn_s_wait_tensorcnt(0); // scalar wait; no-op for non-issuing waves
    __syncthreads();
#else
    uint4 aR[2], bR[2];
    stage_regs(A, Bp, row0, col0, 0, tid, aR, bR);
#pragma unroll
    for (int i = 0; i < 2; ++i) {
        int c = tid + i * 256; int r = c >> 2, kc = (c & 3) * 8;
        *(uint4*)(void*)&sA[0][r][kc] = aR[i];
        *(uint4*)(void*)&sB[0][r][kc] = bR[i];
    }
    __syncthreads();
#endif

    for (int kt = 0; kt < NK; ++kt) {
        const int cur = kt & 1;

#if USE_TDM
        if (tid == 0 && kt + 1 < NK) {   // DMA next tiles into other buffer
            const int nxt = cur ^ 1;
            tdm_load_tile(Abase + (kt + 1) * 32, (unsigned)(uintptr_t)&sA[nxt][0][0]);
            tdm_load_tile(Bbase + (kt + 1) * 32, (unsigned)(uintptr_t)&sB[nxt][0][0]);
        }
#else
        if (kt + 1 < NK)
            stage_regs(A, Bp, row0, col0, (kt + 1) * 32, tid, aR, bR);
        if (kt + 2 < NK) {
            __builtin_prefetch(A  + (row0 + (tid >> 2)) * DD + (kt + 2) * 32, 0, 1);
            __builtin_prefetch(Bp + (col0 + (tid >> 2)) * DD + (kt + 2) * 32, 0, 1);
        }
#endif

        v16bf av[4], bv[2];
#pragma unroll
        for (int mi = 0; mi < 4; ++mi) {
            const int r  = wm * 64 + mi * 16 + (lane & 15);
            const int kb = (lane >> 4) * 8;      // lanes 0-15: K 0-7/16-23, 16-31: 8-15/24-31
            union { uint4 q[2]; v16bf v; } ua;
            ua.q[0] = *(const uint4*)(const void*)&sA[cur][r][kb];
            ua.q[1] = *(const uint4*)(const void*)&sA[cur][r][16 + kb];
            av[mi] = ua.v;
        }
#pragma unroll
        for (int ni = 0; ni < 2; ++ni) {
            const int o  = wn * 32 + ni * 16 + (lane & 15);
            const int kk = (lane >> 4) * 8;      // lanes 0-15: K 0-7/16-23, 16-31: 8-15/24-31
            union { uint4 q[2]; v16bf v; } ub;
            ub.q[0] = *(const uint4*)(const void*)&sB[cur][o][kk];
            ub.q[1] = *(const uint4*)(const void*)&sB[cur][o][16 + kk];
            bv[ni] = ub.v;
        }
#pragma unroll
        for (int mi = 0; mi < 4; ++mi)
#pragma unroll
            for (int ni = 0; ni < 2; ++ni)
                acc[mi][ni] = __builtin_amdgcn_wmma_f32_16x16x32_bf16(
                    false, av[mi], false, bv[ni], (short)0, acc[mi][ni], false, false);

#if USE_TDM
        __builtin_amdgcn_s_wait_tensorcnt(0);   // unconditional scalar wait
#else
        if (kt + 1 < NK) {
            const int nxt = cur ^ 1;
#pragma unroll
            for (int i = 0; i < 2; ++i) {
                int c = tid + i * 256; int r = c >> 2, kc = (c & 3) * 8;
                *(uint4*)(void*)&sA[nxt][r][kc] = aR[i];
                *(uint4*)(void*)&sB[nxt][r][kc] = bR[i];
            }
        }
#endif
        __syncthreads();
    }

    // epilogue: C/D layout — VGPR r: lanes 0-15 -> M=r, lanes 16-31 -> M=8+r
#pragma unroll
    for (int mi = 0; mi < 4; ++mi)
#pragma unroll
        for (int ni = 0; ni < 2; ++ni) {
            const size_t r0 = row0 + wm * 64 + mi * 16 + ((lane >> 4) << 3);
            const size_t c0 = col0 + wn * 32 + ni * 16 + (lane & 15);
            float* cp = Cp + r0 * DD + c0;
#pragma unroll
            for (int r = 0; r < 8; ++r) cp[(size_t)r * DD] = acc[mi][ni][r];
        }
}

// ---------------------------------------------------------------------------
// score/gate: score = (Σh·k − μ_k·Σh) · rsqrt(var_k+eps) / √D  (wave per token)
// ---------------------------------------------------------------------------
__global__ __launch_bounds__(256)
void gate_kernel(const float* __restrict__ kbuf, const __bf16* __restrict__ hn,
                 float* __restrict__ gates) {
    const int tok  = blockIdx.x * 8 + (threadIdx.x >> 5);
    const int lane = threadIdx.x & 31;
    const __bf16* hrow = hn + (size_t)tok * DD;
    const float inv_sqrt_d = 0.03125f;   // 1/sqrt(1024)
    float g = 0.f;
    for (int m = 0; m < BRANCH_; ++m) {
        const float* kr = kbuf + ((size_t)m * NT + tok) * DD;
        float s1 = 0.f, s2 = 0.f, sh = 0.f, sH = 0.f;
#pragma unroll
        for (int j = 0; j < 8; ++j) {
            const int d = j * 128 + lane * 4;
            float4 kv = *(const float4*)(const void*)(kr + d);
            float h0 = (float)hrow[d], h1 = (float)hrow[d + 1];
            float h2 = (float)hrow[d + 2], h3 = (float)hrow[d + 3];
            s1 += kv.x + kv.y + kv.z + kv.w;
            s2 += kv.x * kv.x + kv.y * kv.y + kv.z * kv.z + kv.w * kv.w;
            sh += h0 * kv.x + h1 * kv.y + h2 * kv.z + h3 * kv.w;
            sH += h0 + h1 + h2 + h3;
        }
        s1 = warp_sum(s1); s2 = warp_sum(s2); sh = warp_sum(sh); sH = warp_sum(sH);
        const float mu  = s1 * (1.f / DD);
        const float var = s2 * (1.f / DD) - mu * mu;
        const float rsv = rsqrtf(var + 1e-5f);
        float score = (sh - mu * sH) * rsv * inv_sqrt_d;
        score = fminf(10.f, fmaxf(-10.f, score));
        g += 1.f / (1.f + __expf(-score));
    }
    if (lane == 0) gates[tok] = g * 0.25f;
}

// ---------------------------------------------------------------------------
// fused = Σ_n g_n w_n emb_n / (Σ g_n w_n + 1e-8)  -> bf16
// ---------------------------------------------------------------------------
__global__ __launch_bounds__(256)
void fuse_kernel(const float* __restrict__ tables, const int* __restrict__ buckets,
                 const float* __restrict__ gates, __bf16* __restrict__ fused) {
    const int tok = blockIdx.x;
    float acc[4] = {0.f, 0.f, 0.f, 0.f};
    float gs = 0.f;
    for (int n = 0; n < NGRAM_; ++n) {
        const float g = gates[(size_t)n * NT + tok] * (1.f + 0.3f * n);
        const int bk = buckets[(size_t)n * NT + tok];
        const float* er = tables + ((size_t)n * BUCKET_ + bk) * DD;
#pragma unroll
        for (int i = 0; i < 4; ++i) acc[i] += g * er[threadIdx.x + i * 256];
        gs += g;
    }
    const float inv = 1.f / (gs + 1e-8f);
#pragma unroll
    for (int i = 0; i < 4; ++i)
        fused[(size_t)tok * DD + threadIdx.x + i * 256] = (__bf16)(acc[i] * inv);
}

// ---------------------------------------------------------------------------
// depthwise 3-tap conv over T (+bias), zero pad
// ---------------------------------------------------------------------------
__global__ __launch_bounds__(256)
void conv_kernel(const float* __restrict__ y, const float* __restrict__ w,
                 const float* __restrict__ b, float* __restrict__ out) {
    const int tok = blockIdx.x;
    const int t = tok & (TT - 1);
    const float* yc = y + (size_t)tok * DD;
#pragma unroll
    for (int i = 0; i < 4; ++i) {
        const int d = threadIdx.x + i * 256;
        const float w0 = w[d * 3], w1 = w[d * 3 + 1], w2 = w[d * 3 + 2];
        float v = w1 * yc[d];
        if (t > 0)      v += w0 * yc[d - DD];
        if (t < TT - 1) v += w2 * yc[d + DD];
        out[(size_t)tok * DD + d] = v + b[d];
    }
}

// ---------------------------------------------------------------------------
extern "C" void kernel_launch(void* const* d_in, const int* in_sizes, int n_in,
                              void* d_out, int out_size, void* d_ws, size_t ws_size,
                              hipStream_t stream) {
    (void)in_sizes; (void)n_in; (void)out_size; (void)ws_size;
    const float* hidden = (const float*)d_in[1];
    const int*   bkts   = (const int*)d_in[2];
    const float* tables = (const float*)d_in[3];
    const float* W_K    = (const float*)d_in[4];
    const float* W_V    = (const float*)d_in[5];
    const float* conv_w = (const float*)d_in[6];
    const float* conv_b = (const float*)d_in[7];
    float* out = (float*)d_out;

    char* w = (char*)d_ws;
    const size_t MB = 1024ull * 1024ull;
    __bf16* hnorm  = (__bf16*)(w);                  // 16 MB
    __bf16* enorm  = (__bf16*)(w + 16 * MB);        // 16 MB (reused per n)
    float*  kbuf   = (float*) (w + 32 * MB);        // 128 MB (reused per n)
    float*  gates  = (float*) (w + 160 * MB);       // 96 KB
    __bf16* fusedb = (__bf16*)(w + 161 * MB);       // 16 MB
    float*  ybuf   = (float*) (w + 178 * MB);       // 32 MB
    __bf16* wkb    = (__bf16*)(w + 210 * MB);       // 24 MB  bf16 W_K
    __bf16* wvb    = (__bf16*)(w + 234 * MB);       // 2 MB   bf16 W_V

    cvt_bf16_kernel<<<(BRANCH_ * NGRAM_ * DD * DD) / 1024, 256, 0, stream>>>(W_K, wkb);
    cvt_bf16_kernel<<<(DD * DD) / 1024, 256, 0, stream>>>(W_V, wvb);
    ln_clip_kernel<<<NT, 256, 0, stream>>>(hidden, hnorm);

    for (int n = 0; n < NGRAM_; ++n) {
        gather_ln_kernel<<<NT, 256, 0, stream>>>(
            tables + (size_t)n * BUCKET_ * DD, bkts + (size_t)n * NT, enorm);
        // k[m] = enorm @ W_K[m,n]^T  for m = 0..3 via blockIdx.z
        gemm_bt_wmma<<<dim3(NT / 128, DD / 128, BRANCH_), 256, 0, stream>>>(
            enorm, wkb + (size_t)n * DD * DD, kbuf,
            (size_t)NGRAM_ * DD * DD, (size_t)NT * DD);
        gate_kernel<<<NT / 8, 256, 0, stream>>>(kbuf, hnorm, gates + (size_t)n * NT);
    }

    fuse_kernel<<<NT, 256, 0, stream>>>(tables, bkts, gates, fusedb);
    gemm_bt_wmma<<<dim3(NT / 128, DD / 128, 1), 256, 0, stream>>>(
        fusedb, wvb, ybuf, 0, 0);
    conv_kernel<<<NT, 256, 0, stream>>>(ybuf, conv_w, conv_b, out);
}
